// Lattice_Transformer_SeqLabel_38053410242949
// MI455X (gfx1250) — compile-verified, matrix operand order checked by reading, not run on hardware
//
#include <hip/hip_runtime.h>
#include <hip/hip_bf16.h>

#define B_  2
#define S_  192
#define H_  768
#define NH_ 8
#define DH_ 96
#define FH_ 3072
#define PEROWS 1025
#define PEPAD  1040

typedef __attribute__((ext_vector_type(16))) __bf16 v16bf;
typedef __attribute__((ext_vector_type(8)))  float  v8f;

__device__ __forceinline__ unsigned short f2b(float f) {
  unsigned int u = __float_as_uint(f);
  u += 0x7fffu + ((u >> 16) & 1u);       // round-to-nearest-even
  return (unsigned short)(u >> 16);
}
__device__ __forceinline__ float b2f(unsigned short u) {
  return __uint_as_float(((unsigned int)u) << 16);
}

// ---------------------------------------------------------------------------
// Batched bf16 GEMM: C[M,N] = A[M,K] * B[K,N] (+bias, optional relu), f32 acc.
// B supplied TRANSPOSED as BT[N,K]. Block tile 128x64x32; 8 waves; each wave
// owns a 32x32 C tile -> 4 v_wmma_f32_16x16x32_bf16 per k-step.
// Staging uses GLOBAL_LOAD_ASYNC_TO_LDS_B128 (ASYNCcnt) with double-buffered
// LDS: the async engine copies tile k+1 while WMMAs consume tile k, so the
// copy never touches VGPRs/VALU. Out-of-range rows/cols are CLAMPED on load
// (epilogue guard provides correctness), keeping staging branch-free.
// ---------------------------------------------------------------------------
__global__ __launch_bounds__(256)
void gemm_bf16_wmma(const unsigned short* __restrict__ A, long long lda,
                    long long sA0, long long sA1,
                    const unsigned short* __restrict__ Bt, long long ldbt,
                    long long sB0, long long sB1,
                    float* __restrict__ Cf, unsigned short* __restrict__ Cb,
                    long long ldc, long long sC0, long long sC1,
                    const float* __restrict__ bias,
                    int M, int N, int K, int batch1, int relu)
{
  __shared__ __align__(16) unsigned short As[2][128][40];
  __shared__ __align__(16) unsigned short Bs[2][64][40];
  const int tid = threadIdx.x;
  const int z0 = blockIdx.z / batch1, z1 = blockIdx.z % batch1;
  const unsigned short* Ab = A  + z0 * sA0 + z1 * sA1;
  const unsigned short* Bb = Bt + z0 * sB0 + z1 * sB1;
  const int bm = blockIdx.y * 128;
  const int bn = blockIdx.x * 64;
  const int wave = tid >> 5, lane = tid & 31;
  const int tm = (wave >> 1) * 32;   // 4 row-groups
  const int tn = (wave & 1) * 32;    // 2 col-groups

  union { v8f v; float e[8]; } acc[2][2];
#pragma unroll
  for (int ri = 0; ri < 2; ++ri)
#pragma unroll
    for (int ci = 0; ci < 2; ++ci)
#pragma unroll
      for (int r = 0; r < 8; ++r) acc[ri][ci].e[r] = 0.0f;

  // branch-free staging coordinates (clamped)
  const int ar0 = tid >> 2,         ac0 = (tid & 3) * 8;
  const int ar1 = (tid + 256) >> 2, ac1 = (tid & 3) * 8;
  const int brr = tid >> 2,         bcc = (tid & 3) * 8;
  const int gra0 = min(bm + ar0, M - 1);
  const int gra1 = min(bm + ar1, M - 1);
  const int gcb  = min(bn + brr, N - 1);
  const unsigned short* agp0 = Ab + (long long)gra0 * lda + ac0;
  const unsigned short* agp1 = Ab + (long long)gra1 * lda + ac1;
  const unsigned short* bgp  = Bb + (long long)gcb  * ldbt + bcc;

  // per-buffer LDS byte addresses (generic ptr low 32 bits == LDS offset)
  unsigned ldsA0[2], ldsA1[2], ldsB[2];
#pragma unroll
  for (int bfi = 0; bfi < 2; ++bfi) {
    ldsA0[bfi] = (unsigned)(size_t)&As[bfi][ar0][ac0];
    ldsA1[bfi] = (unsigned)(size_t)&As[bfi][ar1][ac1];
    ldsB[bfi]  = (unsigned)(size_t)&Bs[bfi][brr][bcc];
  }

#define ISSUE_TILE(K0, BUF) do {                                              \
    unsigned long long ga0_ = (unsigned long long)(size_t)(agp0 + (K0));      \
    unsigned long long ga1_ = (unsigned long long)(size_t)(agp1 + (K0));      \
    unsigned long long gb_  = (unsigned long long)(size_t)(bgp  + (K0));      \
    asm volatile("global_load_async_to_lds_b128 %0, %1, off"                  \
                 :: "v"(ldsA0[BUF]), "v"(ga0_) : "memory");                   \
    asm volatile("global_load_async_to_lds_b128 %0, %1, off"                  \
                 :: "v"(ldsA1[BUF]), "v"(ga1_) : "memory");                   \
    asm volatile("global_load_async_to_lds_b128 %0, %1, off"                  \
                 :: "v"(ldsB[BUF]),  "v"(gb_)  : "memory");                   \
  } while (0)

  // prologue: bring in tile 0
  ISSUE_TILE(0, 0);
  asm volatile("s_wait_asynccnt 0x0" ::: "memory");
  __syncthreads();

  const int l15 = lane & 15;
  const int kb  = (lane & 16) ? 8 : 0;

  int cur = 0;
  for (int k0 = 0; k0 < K; k0 += 32) {
    const int nxt = cur ^ 1;
    const bool more = (k0 + 32) < K;
    if (more) ISSUE_TILE(k0 + 32, nxt);   // overlap copy of next tile

    union { v16bf v; uint4 q[2]; } af[2], bf[2];
#pragma unroll
    for (int ri = 0; ri < 2; ++ri) {
      const int mr = tm + ri * 16 + l15;
      af[ri].q[0] = *(const uint4*)&As[cur][mr][kb];
      af[ri].q[1] = *(const uint4*)&As[cur][mr][kb + 16];
    }
#pragma unroll
    for (int ci = 0; ci < 2; ++ci) {
      const int nc = tn + ci * 16 + l15;
      bf[ci].q[0] = *(const uint4*)&Bs[cur][nc][kb];
      bf[ci].q[1] = *(const uint4*)&Bs[cur][nc][kb + 16];
    }
#pragma unroll
    for (int ri = 0; ri < 2; ++ri)
#pragma unroll
      for (int ci = 0; ci < 2; ++ci)
        acc[ri][ci].v = __builtin_amdgcn_wmma_f32_16x16x32_bf16(
            false, af[ri].v, false, bf[ci].v, (short)0, acc[ri][ci].v,
            false, false);

    if (more) asm volatile("s_wait_asynccnt 0x0" ::: "memory");
    __syncthreads();                       // next tile visible; WAR for nxt
    cur = nxt;
  }
#undef ISSUE_TILE

  // epilogue: C 16x16 f32 layout (lanes 0-15 -> M=r, lanes 16-31 -> M=r+8)
  const int mb = (lane & 16) ? 8 : 0;
#pragma unroll
  for (int ci = 0; ci < 2; ++ci) {
    const int gcol = bn + tn + ci * 16 + l15;
    const float bv = (bias && gcol < N) ? bias[gcol] : 0.0f;
#pragma unroll
    for (int ri = 0; ri < 2; ++ri) {
#pragma unroll
      for (int r = 0; r < 8; ++r) {
        const int grow = bm + tm + ri * 16 + mb + r;
        if (grow < M && gcol < N) {
          float vv = acc[ri][ci].e[r] + bv;
          if (relu) vv = fmaxf(vv, 0.0f);
          long long o = z0 * sC0 + z1 * sC1 + (long long)grow * ldc + gcol;
          if (Cf) Cf[o] = vv;
          if (Cb) Cb[o] = f2b(vv);
        }
      }
    }
  }
}

// ---------------------------------------------------------------------------
__global__ void f32_to_bf16_kernel(const float* __restrict__ src,
                                   unsigned short* __restrict__ dst,
                                   int n_dst, int n_src) {
  int i = blockIdx.x * 256 + threadIdx.x;
  if (i < n_dst) dst[i] = (i < n_src) ? f2b(src[i]) : (unsigned short)0;
}

// dst[N][K] = bf16(src[K][N])  (transpose for GEMM BT operand)
__global__ void f32_to_bf16_t_kernel(const float* __restrict__ src,
                                     unsigned short* __restrict__ dst,
                                     int K, int N) {
  long long i = (long long)blockIdx.x * 256 + threadIdx.x;
  if (i < (long long)K * N) {
    int n = (int)(i / K), k = (int)(i - (long long)n * K);
    dst[i] = f2b(src[(long long)k * N + n]);
  }
}

__global__ void add_embed_kernel(const float* __restrict__ w,
                                 const float* __restrict__ be,
                                 float* __restrict__ xf,
                                 unsigned short* __restrict__ xb, int n) {
  int i = blockIdx.x * 256 + threadIdx.x;
  if (i < n) { float v = w[i] + be[i]; xf[i] = v; xb[i] = f2b(v); }
}

// qu = q + u[n];  qv = q + v[n];  c0[b,n,s] = qv . br[n]
__global__ __launch_bounds__(128)
void qkv_post_kernel(const float* __restrict__ qf, const float* __restrict__ u,
                     const float* __restrict__ v, const float* __restrict__ br,
                     float* __restrict__ quf, unsigned short* __restrict__ qub,
                     float* __restrict__ c0) {
  __shared__ float red[128];
  const int s = blockIdx.x, n = blockIdx.y, b = blockIdx.z;
  const int tid = threadIdx.x;
  float part = 0.0f;
  if (tid < DH_) {
    float qv_ = qf[((long long)(b * S_ + s)) * H_ + n * DH_ + tid];
    float quv = qv_ + u[n * DH_ + tid];
    float qvv = qv_ + v[n * DH_ + tid];
    long long o = (((long long)(b * NH_ + n)) * S_ + s) * DH_ + tid;
    quf[o] = quv; qub[o] = f2b(quv);
    part = qvv * br[n * DH_ + tid];
  }
  red[tid] = part; __syncthreads();
  for (int t = 64; t > 0; t >>= 1) { if (tid < t) red[tid] += red[tid + t]; __syncthreads(); }
  if (tid == 0) c0[(b * NH_ + n) * S_ + s] = red[0];
}

// One wave per (b,q,k): hidden_j = relu(T0[d0]+T1[d1]+T2[d2]+T3[d3]+bf1)_j,
// logits = hidden @ Wf2 + bf2, w4 = softmax(logits). uint4 = 8 bf16 per load.
__global__ __launch_bounds__(256)
void fuse_w4_kernel(const unsigned short* __restrict__ T,
                    const float* __restrict__ bf1, const float* __restrict__ Wf2,
                    const float* __restrict__ bf2,
                    const int* __restrict__ pos_s, const int* __restrict__ pos_e,
                    float* __restrict__ w4) {
  const int lane = threadIdx.x & 31;
  const int wid  = blockIdx.x * 8 + (threadIdx.x >> 5);
  const int b = wid / (S_ * S_);
  const int rem = wid - b * S_ * S_;
  const int q = rem / S_;
  const int k = rem - q * S_;
  const int psq = pos_s[b * S_ + q], peq = pos_e[b * S_ + q];
  const int psk = pos_s[b * S_ + k], pek = pos_e[b * S_ + k];
  const unsigned short* T0 = T + ((long long)(0 * PEPAD + (psq - psk + 512))) * FH_;
  const unsigned short* T1 = T + ((long long)(1 * PEPAD + (psq - pek + 512))) * FH_;
  const unsigned short* T2 = T + ((long long)(2 * PEPAD + (peq - psk + 512))) * FH_;
  const unsigned short* T3 = T + ((long long)(3 * PEPAD + (peq - pek + 512))) * FH_;
  float a0 = 0, a1 = 0, a2 = 0, a3 = 0;
  union U8 { uint4 q; unsigned short s[8]; };
#pragma unroll 2
  for (int it = 0; it < FH_ / (32 * 8); ++it) {   // 12 iters, 8 elems each
    const int j = (it * 32 + lane) * 8;
    U8 t0, t1, t2, t3;
    t0.q = *(const uint4*)(T0 + j);
    t1.q = *(const uint4*)(T1 + j);
    t2.q = *(const uint4*)(T2 + j);
    t3.q = *(const uint4*)(T3 + j);
    const float* bj = bf1 + j;
    const float4* wv4 = ((const float4*)Wf2) + j;
#pragma unroll
    for (int e = 0; e < 8; ++e) {
      float h = b2f(t0.s[e]) + b2f(t1.s[e]) + b2f(t2.s[e]) + b2f(t3.s[e]) + bj[e];
      h = fmaxf(h, 0.0f);
      float4 wv = wv4[e];
      a0 += h * wv.x; a1 += h * wv.y; a2 += h * wv.z; a3 += h * wv.w;
    }
  }
#pragma unroll
  for (int off = 16; off > 0; off >>= 1) {
    a0 += __shfl_down(a0, off, 32);
    a1 += __shfl_down(a1, off, 32);
    a2 += __shfl_down(a2, off, 32);
    a3 += __shfl_down(a3, off, 32);
  }
  if (lane == 0) {
    a0 += bf2[0]; a1 += bf2[1]; a2 += bf2[2]; a3 += bf2[3];
    float m = fmaxf(fmaxf(a0, a1), fmaxf(a2, a3));
    float e0 = expf(a0 - m), e1 = expf(a1 - m), e2 = expf(a2 - m), e3 = expf(a3 - m);
    float inv = 1.0f / (e0 + e1 + e2 + e3);
    ((float4*)w4)[wid] = make_float4(e0 * inv, e1 * inv, e2 * inv, e3 * inv);
  }
}

// scores = ((qu.k) + sum_i w4_i*Pq[d_i] + c0)/sqrt(DH); mask; softmax; ctx = p@v
__global__ __launch_bounds__(256)
void attn_kernel(const float* __restrict__ kf, const float* __restrict__ vf,
                 const float* __restrict__ quf, const float* __restrict__ c0,
                 const float* __restrict__ Pq, const float* __restrict__ w4,
                 const int* __restrict__ pos_s, const int* __restrict__ pos_e,
                 const int* __restrict__ seq_len, const int* __restrict__ lex_num,
                 float* __restrict__ ctxf, unsigned short* __restrict__ ctxb) {
  __shared__ float qul[DH_];
  __shared__ float sc[S_];
  __shared__ float red[256];
  const int q = blockIdx.x, n = blockIdx.y, b = blockIdx.z;
  const int tid = threadIdx.x;
  if (tid < DH_)
    qul[tid] = quf[(((long long)(b * NH_ + n)) * S_ + q) * DH_ + tid];
  __syncthreads();
  const int lim = seq_len[b] + lex_num[b];
  const float c0v = c0[(b * NH_ + n) * S_ + q];
  const int psq = pos_s[b * S_ + q], peq = pos_e[b * S_ + q];
  if (tid < S_) {
    const int kk = tid;
    const float4* kp4 = (const float4*)(kf + ((long long)(b * S_ + kk)) * H_ + n * DH_);
    float ac = 0.0f;
#pragma unroll
    for (int d4 = 0; d4 < DH_ / 4; ++d4) {
      float4 kv = kp4[d4];
      ac += qul[d4 * 4 + 0] * kv.x + qul[d4 * 4 + 1] * kv.y +
            qul[d4 * 4 + 2] * kv.z + qul[d4 * 4 + 3] * kv.w;
    }
    const int psk = pos_s[b * S_ + kk], pek = pos_e[b * S_ + kk];
    const float* Pr = Pq + (((long long)(b * NH_ + n)) * S_ + q) * PEPAD;
    const float4 w = ((const float4*)w4)[(b * S_ + q) * S_ + kk];
    float bd = w.x * Pr[psq - psk + 512] + w.y * Pr[psq - pek + 512] +
               w.z * Pr[peq - psk + 512] + w.w * Pr[peq - pek + 512] + c0v;
    float s = (ac + bd) * 0.1020620726f;  // 1/sqrt(96)
    if (kk >= lim) s = -1e15f;
    sc[kk] = s;
  }
  __syncthreads();
  red[tid] = (tid < S_) ? sc[tid] : -3.4e38f;
  __syncthreads();
  for (int t = 128; t > 0; t >>= 1) { if (tid < t) red[tid] = fmaxf(red[tid], red[tid + t]); __syncthreads(); }
  const float mx = red[0];
  __syncthreads();
  float ev = 0.0f;
  if (tid < S_) { ev = expf(sc[tid] - mx); sc[tid] = ev; }
  red[tid] = ev;
  __syncthreads();
  for (int t = 128; t > 0; t >>= 1) { if (tid < t) red[tid] += red[tid + t]; __syncthreads(); }
  const float inv = 1.0f / red[0];
  if (tid < DH_) {
    float a = 0.0f;
    const float* vp = vf + (long long)b * S_ * H_ + n * DH_ + tid;
    for (int kk = 0; kk < S_; ++kk) a += sc[kk] * vp[(long long)kk * H_];
    long long o = ((long long)(b * S_ + q)) * H_ + n * DH_ + tid;
    float r = a * inv;
    ctxf[o] = r; ctxb[o] = f2b(r);
  }
}

// out = LN(a + b) * g + beta  (one block per row of H_)
__global__ __launch_bounds__(256)
void add_ln_kernel(const float* __restrict__ a, const float* __restrict__ bsrc,
                   const float* __restrict__ g, const float* __restrict__ bt,
                   float* __restrict__ outf, unsigned short* __restrict__ outb) {
  __shared__ float xs[H_];
  __shared__ float red[256];
  const int row = blockIdx.x, tid = threadIdx.x;
  float s = 0.0f;
  for (int i = tid; i < H_; i += 256) {
    float x = a[(long long)row * H_ + i] + bsrc[(long long)row * H_ + i];
    xs[i] = x; s += x;
  }
  red[tid] = s; __syncthreads();
  for (int t = 128; t > 0; t >>= 1) { if (tid < t) red[tid] += red[tid + t]; __syncthreads(); }
  const float mean = red[0] * (1.0f / H_);
  __syncthreads();
  float s2 = 0.0f;
  for (int i = tid; i < H_; i += 256) { float d = xs[i] - mean; s2 += d * d; }
  red[tid] = s2; __syncthreads();
  for (int t = 128; t > 0; t >>= 1) { if (tid < t) red[tid] += red[tid + t]; __syncthreads(); }
  const float rstd = rsqrtf(red[0] * (1.0f / H_) + 1e-5f);
  for (int i = tid; i < H_; i += 256) {
    float y = (xs[i] - mean) * rstd * g[i] + bt[i];
    long long o = (long long)row * H_ + i;
    outf[o] = y;
    if (outb) outb[o] = f2b(y);
  }
}

// ---------------------------------------------------------------------------
extern "C" void kernel_launch(void* const* d_in, const int* in_sizes, int n_in,
                              void* d_out, int out_size, void* d_ws, size_t ws_size,
                              hipStream_t stream) {
  (void)in_sizes; (void)n_in; (void)out_size; (void)ws_size;
  const float* word = (const float*)d_in[0];
  const float* bert = (const float*)d_in[1];
  const int* seq_len = (const int*)d_in[2];
  const int* lex_num = (const int*)d_in[3];
  const int* pos_s = (const int*)d_in[4];
  const int* pos_e = (const int*)d_in[5];
  const float* pe  = (const float*)d_in[6];
  const float* Wf1 = (const float*)d_in[7];  const float* bf1 = (const float*)d_in[8];
  const float* Wf2 = (const float*)d_in[9];  const float* bf2 = (const float*)d_in[10];
  const float* Wq  = (const float*)d_in[11]; const float* bq  = (const float*)d_in[12];
  const float* Wk  = (const float*)d_in[13]; const float* bk  = (const float*)d_in[14];
  const float* Wv  = (const float*)d_in[15]; const float* bv  = (const float*)d_in[16];
  const float* Wr  = (const float*)d_in[17]; const float* br  = (const float*)d_in[18];
  const float* u   = (const float*)d_in[19]; const float* v   = (const float*)d_in[20];
  const float* Wo  = (const float*)d_in[21]; const float* bo  = (const float*)d_in[22];
  const float* ln1g = (const float*)d_in[23]; const float* ln1b = (const float*)d_in[24];
  const float* W1  = (const float*)d_in[25]; const float* b1  = (const float*)d_in[26];
  const float* W2  = (const float*)d_in[27]; const float* b2  = (const float*)d_in[28];
  const float* ln2g = (const float*)d_in[29]; const float* ln2b = (const float*)d_in[30];
  float* out = (float*)d_out;

  char* base = (char*)d_ws;
  size_t off = 0;
  auto alloc = [&](size_t bytes) -> void* {
    off = (off + 255) & ~(size_t)255;
    void* p = base + off;
    off += bytes;
    return p;
  };
  const int BS  = B_ * S_;      // 384
  const int BSH = BS * H_;      // 294912

  unsigned short* xb   = (unsigned short*)alloc((size_t)BSH * 2);
  float*          xf   = (float*)alloc((size_t)BSH * 4);
  unsigned short* peb  = (unsigned short*)alloc((size_t)PEPAD * H_ * 2);
  unsigned short* Wf1t = (unsigned short*)alloc((size_t)FH_ * FH_ * 2);
  unsigned short* Wqt  = (unsigned short*)alloc((size_t)H_ * H_ * 2);
  unsigned short* Wkt  = (unsigned short*)alloc((size_t)H_ * H_ * 2);
  unsigned short* Wvt  = (unsigned short*)alloc((size_t)H_ * H_ * 2);
  unsigned short* Wrt  = (unsigned short*)alloc((size_t)H_ * H_ * 2);
  unsigned short* Wot  = (unsigned short*)alloc((size_t)H_ * H_ * 2);
  unsigned short* W1t  = (unsigned short*)alloc((size_t)H_ * H_ * 2);
  unsigned short* W2t  = (unsigned short*)alloc((size_t)H_ * H_ * 2);
  unsigned short* Tb   = (unsigned short*)alloc((size_t)4 * PEPAD * FH_ * 2);
  unsigned short* Ub   = (unsigned short*)alloc((size_t)PEPAD * H_ * 2);
  float* qfb = (float*)alloc((size_t)BSH * 4);
  float* kfb = (float*)alloc((size_t)BSH * 4);
  float* vfb = (float*)alloc((size_t)BSH * 4);
  float* quf = (float*)alloc((size_t)BSH * 4);
  unsigned short* qub = (unsigned short*)alloc((size_t)BSH * 2);
  float* c0  = (float*)alloc((size_t)B_ * NH_ * S_ * 4);
  float* Pq  = (float*)alloc((size_t)B_ * NH_ * S_ * PEPAD * 4);
  float* w4  = (float*)alloc((size_t)B_ * S_ * S_ * 4 * 4);
  float* ctxf = (float*)alloc((size_t)BSH * 4);
  unsigned short* ctxb = (unsigned short*)alloc((size_t)BSH * 2);
  float* aof = (float*)alloc((size_t)BSH * 4);
  float* x1f = (float*)alloc((size_t)BSH * 4);
  unsigned short* x1b = (unsigned short*)alloc((size_t)BSH * 2);
  unsigned short* h1b = (unsigned short*)alloc((size_t)BSH * 2);
  float* h2f = (float*)alloc((size_t)BSH * 4);

  auto cdiv = [](int a, int b) { return (a + b - 1) / b; };

  // weight conversions: bf16 + transpose (GEMM consumes BT[N][K])
  f32_to_bf16_t_kernel<<<cdiv(FH_ * FH_, 256), 256, 0, stream>>>(Wf1, Wf1t, FH_, FH_);
  f32_to_bf16_t_kernel<<<cdiv(H_ * H_, 256), 256, 0, stream>>>(Wq, Wqt, H_, H_);
  f32_to_bf16_t_kernel<<<cdiv(H_ * H_, 256), 256, 0, stream>>>(Wk, Wkt, H_, H_);
  f32_to_bf16_t_kernel<<<cdiv(H_ * H_, 256), 256, 0, stream>>>(Wv, Wvt, H_, H_);
  f32_to_bf16_t_kernel<<<cdiv(H_ * H_, 256), 256, 0, stream>>>(Wr, Wrt, H_, H_);
  f32_to_bf16_t_kernel<<<cdiv(H_ * H_, 256), 256, 0, stream>>>(Wo, Wot, H_, H_);
  f32_to_bf16_t_kernel<<<cdiv(H_ * H_, 256), 256, 0, stream>>>(W1, W1t, H_, H_);
  f32_to_bf16_t_kernel<<<cdiv(H_ * H_, 256), 256, 0, stream>>>(W2, W2t, H_, H_);
  f32_to_bf16_kernel<<<cdiv(PEPAD * H_, 256), 256, 0, stream>>>(pe, peb, PEPAD * H_, PEROWS * H_);

  add_embed_kernel<<<cdiv(BSH, 256), 256, 0, stream>>>(word, bert, xf, xb, BSH);

#define GEMM(Ap, lda, sA0, sA1, Bp, ldbt, sB0, sB1, Cfp, Cbp, ldc, sC0, sC1, biasp, M, N, K, b0, b1v, relu) \
  gemm_bf16_wmma<<<dim3(cdiv((N), 64), cdiv((M), 128), (b0) * (b1v)), 256, 0, stream>>>( \
      Ap, (long long)(lda), (long long)(sA0), (long long)(sA1), \
      Bp, (long long)(ldbt), (long long)(sB0), (long long)(sB1), \
      Cfp, Cbp, (long long)(ldc), (long long)(sC0), (long long)(sC1), \
      biasp, M, N, K, b1v, relu)

  // T_i = pe @ Wf1[i*H:(i+1)*H, :]   batch i=0..3; BT row n, k-offset i*H
  GEMM(peb, H_, 0, 0, Wf1t, FH_, (long long)H_, 0,
       (float*)nullptr, Tb, FH_, (long long)PEPAD * FH_, 0,
       (const float*)nullptr, PEPAD, FH_, H_, 4, 1, 0);
  // U = pe @ Wr -> bf16 [PEPAD][H]  (doubles as BT for the Pq GEMM)
  GEMM(peb, H_, 0, 0, Wrt, H_, 0, 0, (float*)nullptr, Ub, H_, 0, 0,
       (const float*)nullptr, PEPAD, H_, H_, 1, 1, 0);

  // q, k, v projections
  GEMM(xb, H_, 0, 0, Wqt, H_, 0, 0, qfb, (unsigned short*)nullptr, H_, 0, 0, bq, BS, H_, H_, 1, 1, 0);
  GEMM(xb, H_, 0, 0, Wkt, H_, 0, 0, kfb, (unsigned short*)nullptr, H_, 0, 0, bk, BS, H_, H_, 1, 1, 0);
  GEMM(xb, H_, 0, 0, Wvt, H_, 0, 0, vfb, (unsigned short*)nullptr, H_, 0, 0, bv, BS, H_, H_, 1, 1, 0);

  qkv_post_kernel<<<dim3(S_, NH_, B_), 128, 0, stream>>>(qfb, u, v, br, quf, qub, c0);

  // Pq[b,n,q,p] = (q+u).U[p]_n : A = qu[b,n] (S x DH), BT = Ub rows p, k-off n*DH
  GEMM(qub, DH_, (long long)NH_ * S_ * DH_, (long long)S_ * DH_,
       Ub, H_, 0, (long long)DH_,
       Pq, (unsigned short*)nullptr, PEPAD, (long long)NH_ * S_ * PEPAD, (long long)S_ * PEPAD,
       (const float*)nullptr, S_, PEPAD, DH_, B_, NH_, 0);

  fuse_w4_kernel<<<B_ * S_ * S_ / 8, 256, 0, stream>>>(Tb, bf1, Wf2, bf2, pos_s, pos_e, w4);

  attn_kernel<<<dim3(S_, NH_, B_), 256, 0, stream>>>(kfb, vfb, quf, c0, Pq, w4,
                                                     pos_s, pos_e, seq_len, lex_num,
                                                     ctxf, ctxb);

  // ctx @ Wo + bo
  GEMM(ctxb, H_, 0, 0, Wot, H_, 0, 0, aof, (unsigned short*)nullptr, H_, 0, 0, bo, BS, H_, H_, 1, 1, 0);
  add_ln_kernel<<<BS, 256, 0, stream>>>(xf, aof, ln1g, ln1b, x1f, x1b);

  // FFN
  GEMM(x1b, H_, 0, 0, W1t, H_, 0, 0, (float*)nullptr, h1b, H_, 0, 0, b1, BS, H_, H_, 1, 1, 1);
  GEMM(h1b, H_, 0, 0, W2t, H_, 0, 0, h2f, (unsigned short*)nullptr, H_, 0, 0, b2, BS, H_, H_, 1, 1, 0);
  add_ln_kernel<<<BS, 256, 0, stream>>>(x1f, h2f, ln2g, ln2b, out, (unsigned short*)nullptr);

#undef GEMM
}
